// EdgeConv_35991825940497
// MI455X (gfx1250) — matrix operand order repaired and checked by reading, not compiled, not run on hardware
//
#include <hip/hip_runtime.h>
#include <hip/hip_fp16.h>
#include <math.h>

typedef __attribute__((ext_vector_type(16))) _Float16 v16h;
typedef __attribute__((ext_vector_type(8)))  float    v8f;

#define BB 8
#define PP 4096
#define CC 6
#define KK 16
#define HH 64
#define OO 128
#define NROWS (BB * PP * KK)      // 524288 edges
#define NPTS  (BB * PP)           // 32768 points
#define EPSN  1e-5f

#define G1_BLOCKS 1024            // 8 waves * 4 point-tiles each
#define G2_BLOCKS 1024

// ---------------------------------------------------------------------------
// Kernel 1: KNN. One block = 256 points of one batch; whole batch xyz in LDS.
// ---------------------------------------------------------------------------
__global__ __launch_bounds__(256) void knn_kernel(const float* __restrict__ x,
                                                  int* __restrict__ idx) {
    __shared__ float sx[PP], sy[PP], sz[PP];
    const int b = blockIdx.y;
    const float* xb = x + (size_t)b * PP * CC;
    for (int i = threadIdx.x; i < PP; i += 256) {
        sx[i] = xb[i * CC + 0];
        sy[i] = xb[i * CC + 1];
        sz[i] = xb[i * CC + 2];
    }
    __syncthreads();

    const int p = blockIdx.x * 256 + threadIdx.x;
    const float px = sx[p], py = sy[p], pz = sz[p];

    float bd[KK];
    int   bi[KK];
#pragma unroll
    for (int j = 0; j < KK; ++j) { bd[j] = 3.4e38f; bi[j] = 0; }

    for (int q = 0; q < PP; ++q) {
        float dx = sx[q] - px, dy = sy[q] - py, dz = sz[q] - pz;
        float d2 = dx * dx + dy * dy + dz * dz;
        if (d2 < bd[KK - 1]) {
            float cd = d2; int ci = q;
#pragma unroll
            for (int j = 0; j < KK; ++j) {
                if (cd < bd[j]) {
                    float td = bd[j]; int ti = bi[j];
                    bd[j] = cd; bi[j] = ci;
                    cd = td; ci = ti;
                }
            }
        }
    }
    int* op = idx + ((size_t)b * PP + p) * KK;
#pragma unroll
    for (int j = 0; j < KK; ++j) op[j] = bi[j];
}

// ---------------------------------------------------------------------------
// Kernel 2: GEMM1 (edge[12] x W1^T -> 64) via v_wmma_f32_16x16x32_f16.
// One 16-row M-tile == one point's 16 edges. Stores h1 pre-act as f16 and
// produces deterministic per-block partial sums for BN1 stats.
// ---------------------------------------------------------------------------
__global__ __launch_bounds__(256) void gemm1_kernel(const float* __restrict__ x,
                                                    const int* __restrict__ idx,
                                                    const float* __restrict__ W1,
                                                    _Float16* __restrict__ h1,
                                                    float* __restrict__ part1) {
    __shared__ float s_part[8][2 * HH];   // per-wave partials: [sum(64) | sq(64)]

    const int wave = threadIdx.x >> 5;
    const int lane = threadIdx.x & 31;
    const int lr   = lane & 15;           // row-in-tile / col-in-tile index
    const int hi   = lane >> 4;           // which K/M half this lane covers

    // B tiles: W1 as f16, 32x16 per N-tile (K=12 real, rest zero). 4 tiles for H=64.
    v16h Bt[4];
#pragma unroll
    for (int t = 0; t < 4; ++t) {
        v16h bv = {};
        if (hi == 0) {                    // lanes 0-15 hold K=0..15; K>=16 lanes all zero
            const float* wr = W1 + (t * 16 + lr) * (2 * CC);
#pragma unroll
            for (int c = 0; c < 12; ++c) bv[c] = (_Float16)wr[c];
        }
        Bt[t] = bv;
    }

    float accS[4] = {0.f, 0.f, 0.f, 0.f};
    float accQ[4] = {0.f, 0.f, 0.f, 0.f};

    const int tile0 = (blockIdx.x * 8 + wave) * 4;   // 4 point-tiles per wave
    for (int rt = 0; rt < 4; ++rt) {
        const int ptile = tile0 + rt;                // global point index
        const int b = ptile >> 12;
        const int p = ptile & (PP - 1);
        const size_t rowbase = (size_t)ptile * KK;

        // Build A (16x32 f16, K=0..11 real): edge k == row lr of this tile.
        const int j = idx[rowbase + lr];
        const float* xbp = x + (size_t)b * PP * CC;
        const float* xi  = xbp + (size_t)p * CC;
        const float* xj  = xbp + (size_t)j * CC;
        float e[12];
#pragma unroll
        for (int c = 0; c < 6; ++c) { e[c] = xi[c]; e[6 + c] = xj[c] - xi[c]; }

        v16h A = {};
        if (hi == 0) {                    // K = 0..11
#pragma unroll
            for (int c = 0; c < 12; ++c) A[c] = (_Float16)e[c];
        }

        _Float16* hbase = h1 + rowbase * HH;
#pragma unroll
        for (int t = 0; t < 4; ++t) {
            v8f Cc = {};
            Cc = __builtin_amdgcn_wmma_f32_16x16x32_f16(false, A, false, Bt[t],
                                                        (short)0, Cc, false, false);
            float ps = 0.f, pq = 0.f;
#pragma unroll
            for (int r = 0; r < 8; ++r) {
                const float v = Cc[r];
                ps += v; pq += v * v;
                const int m = hi * 8 + r;             // row in tile
                hbase[(size_t)m * HH + t * 16 + lr] = (_Float16)v;
            }
            ps += __shfl_xor(ps, 16, 32);             // fold lane<->lane+16 (same col)
            pq += __shfl_xor(pq, 16, 32);
            accS[t] += ps;
            accQ[t] += pq;
        }
    }

    if (hi == 0) {
#pragma unroll
        for (int t = 0; t < 4; ++t) {
            s_part[wave][t * 16 + lr]      = accS[t];
            s_part[wave][HH + t * 16 + lr] = accQ[t];
        }
    }
    __syncthreads();
    for (int i = threadIdx.x; i < 2 * HH; i += 256) {
        float s = 0.f;
#pragma unroll
        for (int w = 0; w < 8; ++w) s += s_part[w][i];
        part1[(size_t)blockIdx.x * (2 * HH) + i] = s;
    }
}

// ---------------------------------------------------------------------------
// Kernel 3/5: fold partial sums into BN scale/shift (deterministic order).
// ---------------------------------------------------------------------------
__global__ void finalize_kernel(const float* __restrict__ part, int nblocks, int nch,
                                const float* __restrict__ g, const float* __restrict__ be,
                                float* __restrict__ scale, float* __restrict__ shift,
                                float invN) {
    const int c = threadIdx.x;
    if (c >= nch) return;
    float s = 0.f, q = 0.f;
    for (int i = 0; i < nblocks; ++i) {
        s += part[(size_t)i * (2 * nch) + c];
        q += part[(size_t)i * (2 * nch) + nch + c];
    }
    const float mu  = s * invN;
    const float var = q * invN - mu * mu;
    const float rs  = rsqrtf(var + EPSN);
    const float sc  = g[c] * rs;
    scale[c] = sc;
    shift[c] = be[c] - mu * sc;
}

__device__ __forceinline__ float gelu_exact(float v) {
    return 0.5f * v * (1.0f + erff(v * 0.70710678118654752f));
}

// ---------------------------------------------------------------------------
// Kernel 4/6: GEMM2 (64 -> 128) via 2x v_wmma_f32_16x16x32_f16 per N-tile.
// mode 0: accumulate BN2 stats (no h2 materialization).
// mode 1: BN2 + GELU + max over K (M-rows of the tile) -> d_out.
// ---------------------------------------------------------------------------
__global__ __launch_bounds__(256) void gemm2_kernel(const _Float16* __restrict__ h1,
                                                    const float* __restrict__ W2,
                                                    const float* __restrict__ scale1,
                                                    const float* __restrict__ shift1,
                                                    const float* __restrict__ scale2,
                                                    const float* __restrict__ shift2,
                                                    float* __restrict__ part2,
                                                    float* __restrict__ out,
                                                    int mode) {
    __shared__ _Float16 sW2[OO * HH];         // 16 KB of the 320 KB LDS
    __shared__ float s_sc1[HH], s_sh1[HH];
    __shared__ float s_part[8][2 * OO];

    for (int i = threadIdx.x; i < OO * HH; i += 256) sW2[i] = (_Float16)W2[i];
    for (int i = threadIdx.x; i < HH; i += 256) { s_sc1[i] = scale1[i]; s_sh1[i] = shift1[i]; }
    __syncthreads();

    const int wave = threadIdx.x >> 5;
    const int lane = threadIdx.x & 31;
    const int lr   = lane & 15;
    const int hi   = lane >> 4;

    float accS[8], accQ[8];
#pragma unroll
    for (int t = 0; t < 8; ++t) { accS[t] = 0.f; accQ[t] = 0.f; }

    const int p0 = (blockIdx.x * 8 + wave) * 4;
    for (int pt = 0; pt < 4; ++pt) {
        const int ptile = p0 + pt;
        const _Float16* hrow = h1 + (size_t)ptile * KK * HH + (size_t)lr * HH;
        if (ptile + 1 < NPTS)
            __builtin_prefetch(h1 + (size_t)(ptile + 1) * KK * HH + (size_t)lr * HH, 0, 0);

        // A tiles (16x32 each, two chunks cover H=64); apply BN1+GELU on the fly.
        v16h A0 = {}, A1 = {};
#pragma unroll
        for (int gsel = 0; gsel < 4; ++gsel) {
            const int hbase = gsel * 16 + hi * 8;     // 0/16/32/48 (+8 for hi lanes)
#pragma unroll
            for (int c = 0; c < 8; ++c) {
                const int hh = hbase + c;
                const float v  = (float)hrow[hh];
                const float hn = v * s_sc1[hh] + s_sh1[hh];
                const _Float16 ah = (_Float16)gelu_exact(hn);
                if (gsel < 2) A0[gsel * 8 + c] = ah;
                else          A1[(gsel - 2) * 8 + c] = ah;
            }
        }

        float* outp = out + (size_t)ptile * OO;
#pragma unroll
        for (int t = 0; t < 8; ++t) {
            // B tile t, chunks 0/1: lane<16 holds K=0..7,16..23,
            // lane>=16 holds K=8..15,24..31.
            const _Float16* wrow = &sW2[(size_t)(t * 16 + lr) * HH + hi * 8];
            v16h B0, B1;
#pragma unroll
            for (int j = 0; j < 16; ++j) { B0[j] = wrow[j + (j & 8)]; B1[j] = wrow[32 + j + (j & 8)]; }

            v8f Cc = {};
            Cc = __builtin_amdgcn_wmma_f32_16x16x32_f16(false, A0, false, B0,
                                                        (short)0, Cc, false, false);
            Cc = __builtin_amdgcn_wmma_f32_16x16x32_f16(false, A1, false, B1,
                                                        (short)0, Cc, false, false);
            if (mode == 0) {
                float ps = 0.f, pq = 0.f;
#pragma unroll
                for (int r = 0; r < 8; ++r) { const float v = Cc[r]; ps += v; pq += v * v; }
                ps += __shfl_xor(ps, 16, 32);
                pq += __shfl_xor(pq, 16, 32);
                accS[t] += ps;
                accQ[t] += pq;
            } else {
                const int o = t * 16 + lr;
                const float sc = scale2[o], sh = shift2[o];
                float mx = -3.4e38f;
#pragma unroll
                for (int r = 0; r < 8; ++r)
                    mx = fmaxf(mx, gelu_exact(Cc[r] * sc + sh));
                mx = fmaxf(mx, __shfl_xor(mx, 16, 32));   // rows 0-7 vs 8-15
                if (hi == 0) outp[o] = mx;                 // max over K done
            }
        }
    }

    if (mode == 0) {
        if (hi == 0) {
#pragma unroll
            for (int t = 0; t < 8; ++t) {
                s_part[wave][t * 16 + lr]      = accS[t];
                s_part[wave][OO + t * 16 + lr] = accQ[t];
            }
        }
        __syncthreads();
        for (int i = threadIdx.x; i < 2 * OO; i += 256) {
            float s = 0.f;
#pragma unroll
            for (int w = 0; w < 8; ++w) s += s_part[w][i];
            part2[(size_t)blockIdx.x * (2 * OO) + i] = s;
        }
    }
}

// ---------------------------------------------------------------------------
// Host launcher
// ---------------------------------------------------------------------------
extern "C" void kernel_launch(void* const* d_in, const int* in_sizes, int n_in,
                              void* d_out, int out_size, void* d_ws, size_t ws_size,
                              hipStream_t stream) {
    const float* x  = (const float*)d_in[0];
    const float* W1 = (const float*)d_in[1];
    const float* g1 = (const float*)d_in[2];
    const float* b1 = (const float*)d_in[3];
    const float* W2 = (const float*)d_in[4];
    const float* g2 = (const float*)d_in[5];
    const float* b2 = (const float*)d_in[6];
    float* out = (float*)d_out;

    char* ws = (char*)d_ws;
    size_t off = 0;
    _Float16* h1 = (_Float16*)(ws + off); off += (size_t)NROWS * HH * sizeof(_Float16); // 64 MiB
    int*   idx    = (int*)(ws + off);     off += (size_t)NROWS * sizeof(int);           // 2 MiB
    float* part1  = (float*)(ws + off);   off += (size_t)G1_BLOCKS * 2 * HH * sizeof(float);
    float* part2  = (float*)(ws + off);   off += (size_t)G2_BLOCKS * 2 * OO * sizeof(float);
    float* scale1 = (float*)(ws + off);   off += HH * sizeof(float);
    float* shift1 = (float*)(ws + off);   off += HH * sizeof(float);
    float* scale2 = (float*)(ws + off);   off += OO * sizeof(float);
    float* shift2 = (float*)(ws + off);   off += OO * sizeof(float);

    const float invN = 1.0f / (float)NROWS;

    dim3 gknn(PP / 256, BB);
    knn_kernel<<<gknn, 256, 0, stream>>>(x, idx);

    gemm1_kernel<<<G1_BLOCKS, 256, 0, stream>>>(x, idx, W1, h1, part1);
    finalize_kernel<<<1, HH, 0, stream>>>(part1, G1_BLOCKS, HH, g1, b1, scale1, shift1, invN);

    gemm2_kernel<<<G2_BLOCKS, 256, 0, stream>>>(h1, W2, scale1, shift1,
                                                scale2, shift2, part2, out, 0);
    finalize_kernel<<<1, OO, 0, stream>>>(part2, G2_BLOCKS, OO, g2, b2, scale2, shift2, invN);

    gemm2_kernel<<<G2_BLOCKS, 256, 0, stream>>>(h1, W2, scale1, shift1,
                                                scale2, shift2, part2, out, 1);
}
